// Counting_25065429139984
// MI455X (gfx1250) — compile-verified, matrix operand order checked by reading, not run on hardware
//
#include <hip/hip_runtime.h>
#include <hip/hip_bf16.h>

// ---------------------------------------------------------------------------
// Problem constants (reference: B=8, N=2048, D=1024)
// ---------------------------------------------------------------------------
constexpr int Bb = 8;
constexpr int Nn = 2048;
constexpr int Dd = 1024;

typedef __attribute__((ext_vector_type(16))) _Float16 v16h;
typedef __attribute__((ext_vector_type(8)))  _Float16 v8h;
typedef __attribute__((ext_vector_type(4)))  _Float16 v4h;
typedef __attribute__((ext_vector_type(8)))  float    v8f;
typedef __attribute__((ext_vector_type(4)))  float    v4f;

// Load a 16x32 f16 WMMA A-fragment (or 32x16 B-fragment of a row-major
// transposed matrix) for this lane.  Per CDNA5 ISA 16-bit A layout:
//   lanes 0-15  : row = lane,    K in [0,8) U [16,24)
//   lanes 16-31 : row = lane-16, K in [8,16) U [24,32)
// rowptr points at the fragment row's k=0 element; kbase is the K-chunk base.
__device__ __forceinline__ v16h load_frag(const _Float16* __restrict__ rowptr,
                                          int kbase, int seg) {
  union { v16h v; v8h h[2]; } u;
  u.h[0] = *(const v8h*)(rowptr + kbase + seg * 8);        // 16B load
  u.h[1] = *(const v8h*)(rowptr + kbase + seg * 8 + 16);   // 16B load
  return u.v;
}

__device__ __forceinline__ v16h load_frag_lds(const _Float16* rowptr,
                                              int kbase, int seg) {
  union { v16h v; v8h h[2]; } u;
  u.h[0] = *(const v8h*)(rowptr + kbase + seg * 8);
  u.h[1] = *(const v8h*)(rowptr + kbase + seg * 8 + 16);
  return u.v;
}

// ---------------------------------------------------------------------------
// Kernel 1: per-row L2 norm; emit f16 copies of data and normed.
// One 256-thread block per row; each thread handles 4 consecutive floats.
// data is read exactly once -> non-temporal load.
// ---------------------------------------------------------------------------
__global__ __launch_bounds__(256) void normalize_kernel(
    const float* __restrict__ data,
    _Float16* __restrict__ data_h,
    _Float16* __restrict__ normed_h) {
  __shared__ float red[256];
  const int row = blockIdx.x;           // 0 .. B*N-1
  const int t   = threadIdx.x;
  const v4f v =
      __builtin_nontemporal_load((const v4f*)(data + (size_t)row * Dd) + t);
  float ss = v[0] * v[0] + v[1] * v[1] + v[2] * v[2] + v[3] * v[3];
  red[t] = ss;
  __syncthreads();
  for (int s = 128; s > 0; s >>= 1) {
    if (t < s) red[t] += red[t + s];
    __syncthreads();
  }
  const float scale = rsqrtf(fmaxf(red[0], 1e-12f));
  v4h dh = {(_Float16)v[0], (_Float16)v[1], (_Float16)v[2], (_Float16)v[3]};
  v4h nh = {(_Float16)(v[0] * scale), (_Float16)(v[1] * scale),
            (_Float16)(v[2] * scale), (_Float16)(v[3] * scale)};
  ((v4h*)(data_h   + (size_t)row * Dd))[t] = dh;
  ((v4h*)(normed_h + (size_t)row * Dd))[t] = nh;
}

// ---------------------------------------------------------------------------
// Kernel 2: W2 [2D, D] f32 row-major -> W2T [D, 2D] f16 (transposed),
// so that WMMA B-fragments (one output column per lane, K-major) load as
// contiguous 16-byte chunks.
// ---------------------------------------------------------------------------
__global__ __launch_bounds__(256) void w2t_kernel(
    const float* __restrict__ W2, _Float16* __restrict__ W2T) {
  const int idx = blockIdx.x * 256 + threadIdx.x;    // over D * 2D
  const int n = idx / (2 * Dd);                      // output column
  const int k = idx % (2 * Dd);                      // reduction index
  W2T[idx] = (_Float16)W2[(size_t)k * Dd + n];
}

// ---------------------------------------------------------------------------
// Kernel 3: fused  sum_m relu(normed_n . normed_m)  per row.
// One wave owns a 16-row tile of one batch; A-tile (16x1024 f16 = 32 KB)
// lives in LDS.  Four M-tiles are processed per pass so each A fragment is
// reused 4x and four independent accumulators keep the XDL pipe busy.
// The NxN sim matrix is never materialized.
// ---------------------------------------------------------------------------
__global__ __launch_bounds__(32) void counter_kernel(
    const _Float16* __restrict__ normed, float* __restrict__ cOut) {
  __shared__ __align__(16) _Float16 As[16 * Dd];     // 32 KB
  const int wave = blockIdx.x;                       // B * N/16 waves
  const int b    = wave / (Nn / 16);
  const int rt   = wave % (Nn / 16);
  const int lane = threadIdx.x;
  const int row  = lane & 15;
  const int seg  = lane >> 4;

  // Stage the 16 A rows into LDS (64 x 16B per lane).
  const _Float16* abase = normed + ((size_t)b * Nn + (size_t)rt * 16) * Dd;
  for (int i = lane; i < 16 * Dd / 8; i += 32)
    ((v8h*)As)[i] = ((const v8h*)abase)[i];
  __syncthreads();

  const _Float16* arow = &As[row * Dd];

  float rowacc[8];
#pragma unroll
  for (int j = 0; j < 8; ++j) rowacc[j] = 0.f;

  const _Float16* bbat = normed + ((size_t)b * Nn + row) * Dd;

  for (int mt = 0; mt < Nn / 16; mt += 4) {
    const _Float16* brow[4];
#pragma unroll
    for (int t = 0; t < 4; ++t)
      brow[t] = bbat + (size_t)(mt + t) * 16 * Dd;

    v8f acc[4];
#pragma unroll
    for (int t = 0; t < 4; ++t)
#pragma unroll
      for (int j = 0; j < 8; ++j) acc[t][j] = 0.f;

#pragma unroll 2
    for (int kc = 0; kc < Dd; kc += 32) {
      const v16h a = load_frag_lds(arow, kc, seg);
#pragma unroll
      for (int t = 0; t < 4; ++t) {
        const v16h bm = load_frag(brow[t], kc, seg);
        acc[t] = __builtin_amdgcn_wmma_f32_16x16x32_f16(
            false, a, false, bm, (short)0, acc[t], false, false);
      }
    }
    // relu then fold these tiles' columns into the running row sums
#pragma unroll
    for (int t = 0; t < 4; ++t)
#pragma unroll
      for (int j = 0; j < 8; ++j) rowacc[j] += fmaxf(acc[t][j], 0.f);
  }

  // Reduce over the 16 lanes of each half (N dimension of the D tile).
#pragma unroll
  for (int j = 0; j < 8; ++j) {
    float v = rowacc[j];
    v += __shfl_xor(v, 1, 32);
    v += __shfl_xor(v, 2, 32);
    v += __shfl_xor(v, 4, 32);
    v += __shfl_xor(v, 8, 32);
    rowacc[j] = v;
  }
  // lane 0 holds rows 0..7 of the tile, lane 16 rows 8..15
  if (row == 0) {
#pragma unroll
    for (int j = 0; j < 8; ++j)
      cOut[(size_t)b * Nn + (size_t)rt * 16 + seg * 8 + j] = rowacc[j];
  }
}

// ---------------------------------------------------------------------------
// Kernel 4: counter_h[row, d] = f16( softplus(c[row] * W1[d] + b1[d]) )
// ---------------------------------------------------------------------------
__global__ __launch_bounds__(256) void expand_kernel(
    const float* __restrict__ c, const float* __restrict__ W1,
    const float* __restrict__ b1, _Float16* __restrict__ cnt_h) {
  const int row = blockIdx.x;
  const int t   = threadIdx.x;
  const float cv = c[row];
#pragma unroll
  for (int i = 0; i < 4; ++i) {
    const int d = i * 256 + t;
    const float x  = fmaf(cv, W1[d], b1[d]);
    const float sp = (x > 20.f) ? x : log1pf(__expf(x));
    cnt_h[(size_t)row * Dd + d] = (_Float16)sp;
  }
}

// ---------------------------------------------------------------------------
// Kernel 5: out[BN, D] = Xd @ W2T[:, 0:D]^T + Xc @ W2T[:, D:2D]^T   (fp32 out)
// Each wave computes a 32x64 output tile: 2 A-fragments x 4 B-fragments per
// K-chunk feed 8 WMMAs (0.75 fragment loads per WMMA); B frags are shared by
// both row halves.  K = 2048 split into the data half and the counter half.
// Output written once -> non-temporal stores.
// ---------------------------------------------------------------------------
__global__ __launch_bounds__(256) void gemm_kernel(
    const _Float16* __restrict__ Xd, const _Float16* __restrict__ Xc,
    const _Float16* __restrict__ W2T, float* __restrict__ out) {
  const int lane = threadIdx.x & 31;
  const int w    = (blockIdx.x * 256 + threadIdx.x) >> 5;  // global wave id
  const int rt   = w >> 4;                                 // 512 row tiles (32 rows)
  const int cg   = w & 15;                                 // 16 col groups (64 cols)
  const int col  = lane & 15;
  const int seg  = lane >> 4;
  const size_t rbase = (size_t)rt * 32;
  const int    cbase = cg * 64;

  v8f acc[2][4];
#pragma unroll
  for (int r2 = 0; r2 < 2; ++r2)
#pragma unroll
    for (int t4 = 0; t4 < 4; ++t4)
#pragma unroll
      for (int j = 0; j < 8; ++j) acc[r2][t4][j] = 0.f;

  const _Float16* brows[4];
#pragma unroll
  for (int t4 = 0; t4 < 4; ++t4)
    brows[t4] = W2T + (size_t)(cbase + t4 * 16 + col) * (2 * Dd);

  // K-half 1: original data against W2 rows [0, D)
  {
    const _Float16* ar0 = Xd + (rbase + col) * Dd;
    const _Float16* ar1 = Xd + (rbase + 16 + col) * Dd;
    for (int kc = 0; kc < Dd; kc += 32) {
      const v16h a0 = load_frag(ar0, kc, seg);
      const v16h a1 = load_frag(ar1, kc, seg);
#pragma unroll
      for (int t4 = 0; t4 < 4; ++t4) {
        const v16h bm = load_frag(brows[t4], kc, seg);
        acc[0][t4] = __builtin_amdgcn_wmma_f32_16x16x32_f16(
            false, a0, false, bm, (short)0, acc[0][t4], false, false);
        acc[1][t4] = __builtin_amdgcn_wmma_f32_16x16x32_f16(
            false, a1, false, bm, (short)0, acc[1][t4], false, false);
      }
    }
  }
  // K-half 2: counter matrix against W2 rows [D, 2D)
  {
    const _Float16* ar0 = Xc + (rbase + col) * Dd;
    const _Float16* ar1 = Xc + (rbase + 16 + col) * Dd;
    for (int kc = 0; kc < Dd; kc += 32) {
      const v16h a0 = load_frag(ar0, kc, seg);
      const v16h a1 = load_frag(ar1, kc, seg);
#pragma unroll
      for (int t4 = 0; t4 < 4; ++t4) {
        const v16h bm = load_frag(brows[t4], Dd + kc, seg);
        acc[0][t4] = __builtin_amdgcn_wmma_f32_16x16x32_f16(
            false, a0, false, bm, (short)0, acc[0][t4], false, false);
        acc[1][t4] = __builtin_amdgcn_wmma_f32_16x16x32_f16(
            false, a1, false, bm, (short)0, acc[1][t4], false, false);
      }
    }
  }

  // D tile layout: VGPR j -> (M = j + 8*seg, N = col)
#pragma unroll
  for (int r2 = 0; r2 < 2; ++r2)
#pragma unroll
    for (int t4 = 0; t4 < 4; ++t4)
#pragma unroll
      for (int j = 0; j < 8; ++j)
        __builtin_nontemporal_store(
            acc[r2][t4][j],
            out + (rbase + r2 * 16 + seg * 8 + j) * Dd + cbase + t4 * 16 + col);
}

// ---------------------------------------------------------------------------
// Host-side launcher.  Workspace layout (bytes):
//   [0,   32M)  data_h   f16 [B*N, D]
//   [32M, 64M)  tmp_h    f16 [B*N, D]   (normed, later overwritten by counter)
//   [64M, 68M)  W2T      f16 [D, 2D]
//   [68M, ...)  cbuf     f32 [B*N]
// ---------------------------------------------------------------------------
extern "C" void kernel_launch(void* const* d_in, const int* in_sizes, int n_in,
                              void* d_out, int out_size, void* d_ws,
                              size_t ws_size, hipStream_t stream) {
  const float* data = (const float*)d_in[0];   // [B,N,D]
  const float* W1   = (const float*)d_in[1];   // [1,D]
  const float* b1   = (const float*)d_in[2];   // [D]
  const float* W2   = (const float*)d_in[3];   // [2D,D]
  float* out = (float*)d_out;

  char* ws = (char*)d_ws;
  const size_t szHalf = (size_t)Bb * Nn * Dd * sizeof(_Float16);   // 32 MB
  _Float16* data_h = (_Float16*)(ws);
  _Float16* tmp_h  = (_Float16*)(ws + szHalf);
  _Float16* w2t    = (_Float16*)(ws + 2 * szHalf);
  float*    cbuf   = (float*)   (ws + 2 * szHalf + (size_t)Dd * 2 * Dd * 2);

  // 1. norms + f16 staging
  normalize_kernel<<<Bb * Nn, 256, 0, stream>>>(data, data_h, tmp_h);
  // 2. W2 transpose -> f16
  w2t_kernel<<<(Dd * 2 * Dd) / 256, 256, 0, stream>>>(W2, w2t);
  // 3. fused sim/relu/rowsum (one wave per 16-row tile, 4 M-tiles per pass)
  counter_kernel<<<Bb * Nn / 16, 32, 0, stream>>>(tmp_h, cbuf);
  // 4. softplus expansion (overwrites tmp_h with counter matrix)
  expand_kernel<<<Bb * Nn, 256, 0, stream>>>(cbuf, W1, b1, tmp_h);
  // 5. final merger GEMM (32x64 tile per wave)
  gemm_kernel<<<(Bb * Nn / 32) * (Dd / 64) / 8, 256, 0, stream>>>(
      data_h, tmp_h, w2t, out);
}
